// MambaBlock_74294344286534
// MI455X (gfx1250) — matrix operand order repaired and checked by reading, not compiled, not run on hardware
//
#include <hip/hip_runtime.h>

// ---------------------------------------------------------------------------
// MambaBlock for MI455X (gfx1250): bf16 WMMA GEMMs + persistent WMMA scan,
// with TDM (tensor_load_to_lds) staging of weight tiles into LDS.
// D = 1024, BATCH = 8, SEQ = 4096, M = BATCH*SEQ = 32768.
// ---------------------------------------------------------------------------

#define DIM   1024
#define BATCH 8
#define SEQ   4096
#define MROWS (BATCH * SEQ)

typedef __attribute__((ext_vector_type(16))) __bf16 v16bf;
typedef __attribute__((ext_vector_type(8)))  __bf16 bf16x8;
typedef __attribute__((ext_vector_type(8)))  float  v8f;
typedef __attribute__((ext_vector_type(8)))  float  f32x8;
typedef __attribute__((ext_vector_type(4)))  unsigned int u32x4;
typedef __attribute__((ext_vector_type(8)))  int    i32x8;
typedef __attribute__((ext_vector_type(4)))  int    i32x4;

// TDM pad_enable inserts 16B after every 1024B stored -> effective LDS row
// stride for a 1024-elem bf16 row is 1040 elems, with an 8-elem pad at k=512.
#define LDSTRIDE 1040
// blended-state staging stride (written by VALU, conflict-free padding)
#define KP_BL 1032

// ---------------------------------------------------------------------------
// TDM: DMA a 2-D bf16 tile (numRows x rowLen elems, row-major, row stride
// rowStride elems) from global into LDS at byte offset ldsOff, with hardware
// padding (16B per 1024B) for bank-conflict-free column reads.
// Issue once per workgroup (wave 0); completion via s_wait_tensorcnt.
// This toolchain exposes the 6-arg builtin:
//   (u32x4 g0, i32x8 g1, i32x4 g2, i32x4 g3, i32x8 aux, i32 cpol)
// ---------------------------------------------------------------------------
__device__ __forceinline__ void tdm_load_2d_bf16(unsigned ldsOff, const void* gsrc,
                                                 int rowLen, int numRows,
                                                 int tensorRows, int rowStride) {
    unsigned long long ga = (unsigned long long)(uintptr_t)gsrc;
    u32x4 g0;
    g0[0] = 1u;                                          // count=1 (user D#)
    g0[1] = ldsOff;                                      // lds_addr (bytes)
    g0[2] = (unsigned)(ga & 0xffffffffu);                // global_addr[31:0]
    g0[3] = (unsigned)((ga >> 32) & 0x1ffffffu)          // global_addr[56:32]
          | (2u << 30);                                  // type = 2 ("image")
    i32x8 g1;
    g1[0] = (1 << 16)                                    // data_size = 2 bytes
          | (1 << 20)                                    // pad_enable
          | (7 << 22)                                    // pad_interval: 256 DW
          | (3 << 25);                                   // pad_amount: 4 DW
    g1[1] = (rowLen & 0xffff) << 16;                     // tensor_dim0[15:0]
    g1[2] = ((rowLen >> 16) & 0xffff)                    // tensor_dim0[31:16]
          | ((tensorRows & 0xffff) << 16);               // tensor_dim1[15:0]
    g1[3] = ((tensorRows >> 16) & 0xffff)                // tensor_dim1[31:16]
          | ((rowLen & 0xffff) << 16);                   // tile_dim0
    g1[4] = numRows & 0xffff;                            // tile_dim1 (dim2=0)
    g1[5] = rowStride;                                   // tensor_dim0_stride
    g1[6] = 0;
    g1[7] = 0;
    i32x4 z4 = (i32x4)0;                                 // 2-D: groups 2/3 unused
    i32x8 z8 = (i32x8)0;
    __builtin_amdgcn_tensor_load_to_lds(g0, g1, z4, z4, z8, 0);
}

__device__ __forceinline__ unsigned lds_off(const void* p) {
    return (unsigned)(unsigned long long)(uintptr_t)p;   // low 32 bits = LDS offset
}

// padded LDS k-index: 8-elem pad after first 512 elems of each row
__device__ __forceinline__ int kpad(int k) { return k + ((k >> 9) << 3); }

// ---------------------------------------------------------------------------
// fp32 -> bf16 conversion
// ---------------------------------------------------------------------------
__global__ __launch_bounds__(256) void cvt_f32_bf16(const float* __restrict__ src,
                                                    __bf16* __restrict__ dst, int n) {
    int i = blockIdx.x * 256 + threadIdx.x;
    if (i < n) dst[i] = (__bf16)src[i];
}

// ---------------------------------------------------------------------------
// C(M x N) = act( Xbf16(M x K) @ Wbf16^T + bias ), f32 out (+ optional bf16).
// Block: 256 threads (8 waves). Tile: 128(M) x 64(N); wave w -> M-tile w,
// 4 N-tiles. The 64 x K weight slice is DMA'd into LDS by the TDM.
// ---------------------------------------------------------------------------
__global__ __launch_bounds__(256) void gemm_bf16_wmma(
    const __bf16* __restrict__ X, const __bf16* __restrict__ W,
    const float* __restrict__ bias, float* __restrict__ Cf,
    __bf16* __restrict__ Cb, int M, int N, int K, int act) {
    __shared__ __bf16 wl[64 * LDSTRIDE];

    const int tid  = threadIdx.x;
    const int wave = tid >> 5;
    const int lane = tid & 31;
    const int mblk = blockIdx.x;
    const int nblk = blockIdx.y;

    if (wave == 0) {  // one TDM transfer for the whole block's weight slice
        tdm_load_2d_bf16(lds_off(wl), &W[(size_t)(nblk * 64) * K],
                         K, 64, 64, K);
        __builtin_amdgcn_s_wait_tensorcnt(0);
    }
    __syncthreads();

    v8f acc[4];
    #pragma unroll
    for (int f = 0; f < 4; ++f) acc[f] = (v8f)0.f;

    const int mrow  = mblk * 128 + wave * 16 + (lane & 15);
    const int koff  = (lane >> 4) * 8;       // A-frag K split per ISA layout
    const int nlane = lane & 15;
    const __bf16* xrow = X + (size_t)mrow * K;

    for (int kt = 0; kt < K / 32; ++kt) {
        int k0 = kt * 32 + koff;
        bf16x8 a0 = *(const bf16x8*)(xrow + k0);        // K = k0..k0+7
        bf16x8 a1 = *(const bf16x8*)(xrow + k0 + 16);   // K = k0+16..k0+23
        v16bf a;
        #pragma unroll
        for (int e = 0; e < 8; ++e) { a[e] = a0[e]; a[e + 8] = a1[e]; }

        int kb = kpad(kt * 32 + (lane >> 4) * 8);       // B-frag K split + pad
        #pragma unroll
        for (int f = 0; f < 4; ++f) {
            const __bf16* bp = &wl[(f * 16 + nlane) * LDSTRIDE + kb];
            bf16x8 b0 = *(const bf16x8*)bp;
            bf16x8 b1 = *(const bf16x8*)(bp + 16);
            v16bf b;
            #pragma unroll
            for (int e = 0; e < 8; ++e) { b[e] = b0[e]; b[e + 8] = b1[e]; }
            acc[f] = __builtin_amdgcn_wmma_f32_16x16x32_bf16(
                false, a, false, b, (short)0, acc[f], false, false);
        }
    }

    #pragma unroll
    for (int f = 0; f < 4; ++f) {
        int n = nblk * 64 + f * 16 + nlane;
        float bs = bias ? bias[n] : 0.f;
        #pragma unroll
        for (int v = 0; v < 8; ++v) {
            int m = mblk * 128 + wave * 16 + v + 8 * (lane >> 4);  // C layout
            float val = acc[f][v] + bs;
            if (act == 1) val = 1.f / (1.f + __expf(-val));        // sigmoid
            size_t idx = (size_t)m * N + n;
            Cf[idx] = val;
            if (Cb) Cb[idx] = (__bf16)val;
        }
    }
}

// ---------------------------------------------------------------------------
// Persistent scan: 8 workgroups x 256 threads (8 waves). WG wg owns columns
// [wg*128, wg*128+128); its A^T slice (128 rows x K=1024, bf16) is TDM-loaded
// into LDS once and stays resident for all 4096 steps. Each wave owns one
// 16-column tile. Per step: stage blended_t (8x1024 bf16) into LDS once per
// WG, wmma over K, tanh(+u) epilogue, write new state, atomic grid barrier.
// State is double-buffered in global so one barrier per step suffices.
// ---------------------------------------------------------------------------
__global__ __launch_bounds__(256) void scan_wmma(
    const float* __restrict__ inp, const float* __restrict__ gate,
    const float* __restrict__ u, const __bf16* __restrict__ Abf,
    float* __restrict__ stateBuf,      // 2 * BATCH * DIM fp32, zero-initialized
    __bf16* __restrict__ statesOut,    // (b, s, d) bf16 scan output
    unsigned* __restrict__ bar) {      // zero-initialized grid barrier counter
    __shared__ __bf16 al[128 * LDSTRIDE];   // A^T slice: [localCol][k] (padded)
    __shared__ __bf16 bl[BATCH * KP_BL];    // blended_t: [b][k]

    const int tid  = threadIdx.x;
    const int wave = tid >> 5;
    const int lane = tid & 31;
    const int wg   = blockIdx.x;
    const int NWG  = gridDim.x;

    if (wave == 0) {  // one-time TDM DMA of this WG's 128-row slice of A
        tdm_load_2d_bf16(lds_off(al), &Abf[(size_t)(wg * 128) * DIM],
                         DIM, 128, 128, DIM);
        __builtin_amdgcn_s_wait_tensorcnt(0);
    }
    __syncthreads();

    const int nbase = (wg * 8 + wave) * 16;    // global output-column base
    const int mm    = (lane & 15) & 7;          // batch row (rows 8..15 mirror 0..7)
    const int koff  = (lane >> 4) * 8;
    const int nlane = lane & 15;

    for (int t = 0; t < SEQ; ++t) {
        const float* sb = stateBuf + (size_t)(t & 1) * BATCH * DIM;
        float*       sn = stateBuf + (size_t)((t + 1) & 1) * BATCH * DIM;

        // ---- stage blended_t = g*state + (1-g)*inp into LDS (once per WG) ----
        #pragma unroll
        for (int j = 0; j < 4; ++j) {
            int c = tid + 256 * j;            // 1024 chunks of 8 floats
            int b = c / 128;
            int k = (c & 127) * 8;
            size_t r = (size_t)(b * SEQ + t) * DIM + k;
            f32x8 gv = *(const f32x8*)(gate + r);
            f32x8 iv = *(const f32x8*)(inp + r);
            f32x8 sv = *(const f32x8*)(sb + b * DIM + k);
            f32x8 blv = gv * sv + (1.f - gv) * iv;
            bf16x8 o;
            #pragma unroll
            for (int e = 0; e < 8; ++e) o[e] = (__bf16)blv[e];
            *(bf16x8*)(&bl[b * KP_BL + k]) = o;
        }
        __syncthreads();

        // ---- GEMM: (16x1024 blended, 8 valid rows) @ A^T-slice -> 16x16 tile ----
        v8f acc = (v8f)0.f;
        for (int kt = 0; kt < DIM / 32; ++kt) {
            int k0 = kt * 32 + koff;
            bf16x8 a0 = *(const bf16x8*)(&bl[mm * KP_BL + k0]);
            bf16x8 a1 = *(const bf16x8*)(&bl[mm * KP_BL + k0 + 16]);
            v16bf a;
            #pragma unroll
            for (int e = 0; e < 8; ++e) { a[e] = a0[e]; a[e + 8] = a1[e]; }

            int kb = kpad(kt * 32 + (lane >> 4) * 8);
            const __bf16* bp = &al[(wave * 16 + nlane) * LDSTRIDE + kb];
            bf16x8 b0 = *(const bf16x8*)bp;
            bf16x8 b1 = *(const bf16x8*)(bp + 16);
            v16bf b;
            #pragma unroll
            for (int e = 0; e < 8; ++e) { b[e] = b0[e]; b[e + 8] = b1[e]; }
            acc = __builtin_amdgcn_wmma_f32_16x16x32_bf16(
                false, a, false, b, (short)0, acc, false, false);
        }

        // ---- epilogue: rows 0..7 live in lanes 0..15, VGPR v = batch v ----
        if (lane < 16) {
            int col = nbase + nlane;
            #pragma unroll
            for (int v = 0; v < 8; ++v) {
                size_t r = (size_t)(v * SEQ + t) * DIM + col;
                float s = tanhf(acc[v] + u[r]);
                sn[v * DIM + col] = s;
                statesOut[r] = (__bf16)s;
            }
        }

        // ---- grid barrier (release -> arrive -> spin -> acquire) ----
        __threadfence();
        __syncthreads();
        if (tid == 0) {
            atomicAdd(bar, 1u);
            unsigned target = (unsigned)NWG * (unsigned)(t + 1);
            while (__hip_atomic_load(bar, __ATOMIC_RELAXED,
                                     __HIP_MEMORY_SCOPE_AGENT) < target) {
                __builtin_amdgcn_s_sleep(1);
            }
        }
        __syncthreads();
        __threadfence();
    }
}

// ---------------------------------------------------------------------------
// Host-side orchestration (graph-capture safe: only async launches/memsets).
// ---------------------------------------------------------------------------
extern "C" void kernel_launch(void* const* d_in, const int* in_sizes, int n_in,
                              void* d_out, int out_size, void* d_ws, size_t ws_size,
                              hipStream_t stream) {
    (void)in_sizes; (void)n_in; (void)out_size; (void)ws_size;

    const float* x      = (const float*)d_in[0];
    const float* A      = (const float*)d_in[1];
    const float* Bm     = (const float*)d_in[2];
    const float* W_in   = (const float*)d_in[3];
    const float* b_in   = (const float*)d_in[4];
    const float* W_gate = (const float*)d_in[5];
    const float* b_gate = (const float*)d_in[6];
    const float* W_out  = (const float*)d_in[7];
    const float* b_out  = (const float*)d_in[8];
    float* out = (float*)d_out;

    const size_t MB = (size_t)1 << 20;
    char* w = (char*)d_ws;
    __bf16*   x_bf     = (__bf16*)(w + 0);          //  64 MB: x in bf16
    float*    inp_f    = (float*)(w + 64 * MB);     // 128 MB: in_proj fp32
    __bf16*   inp_b    = (__bf16*)(w + 192 * MB);   //  64 MB: in_proj bf16
    float*    gate_f   = (float*)(w + 256 * MB);    // 128 MB: sigmoid gate fp32
    float*    u_f      = (float*)(w + 384 * MB);    // 128 MB: u = inp @ B^T fp32
    __bf16*   states_b = (__bf16*)(w + 512 * MB);   //  64 MB: scan states bf16
    __bf16*   Wi_b     = (__bf16*)(w + 576 * MB);   //   2 MB each, bf16 weights
    __bf16*   Wg_b     = (__bf16*)(w + 578 * MB);
    __bf16*   Bm_b     = (__bf16*)(w + 580 * MB);
    __bf16*   A_b      = (__bf16*)(w + 582 * MB);
    __bf16*   Wo_b     = (__bf16*)(w + 584 * MB);
    float*    stateBuf = (float*)(w + 586 * MB);    // 64 KB: double-buffered state
    unsigned* bar      = (unsigned*)(w + 587 * MB); // grid barrier counter

    // zero the recurrent state and barrier every launch (ws is not re-poisoned)
    (void)hipMemsetAsync(stateBuf, 0, 2 * BATCH * DIM * sizeof(float), stream);
    (void)hipMemsetAsync(bar, 0, 64, stream);

    // fp32 -> bf16 conversions
    const int NX = MROWS * DIM, NW = DIM * DIM;
    cvt_f32_bf16<<<(NX + 255) / 256, 256, 0, stream>>>(x, x_bf, NX);
    cvt_f32_bf16<<<(NW + 255) / 256, 256, 0, stream>>>(W_in,   Wi_b, NW);
    cvt_f32_bf16<<<(NW + 255) / 256, 256, 0, stream>>>(W_gate, Wg_b, NW);
    cvt_f32_bf16<<<(NW + 255) / 256, 256, 0, stream>>>(Bm,     Bm_b, NW);
    cvt_f32_bf16<<<(NW + 255) / 256, 256, 0, stream>>>(A,      A_b,  NW);
    cvt_f32_bf16<<<(NW + 255) / 256, 256, 0, stream>>>(W_out,  Wo_b, NW);

    dim3 ggrid(MROWS / 128, DIM / 64);
    // inp = x @ W_in^T + b_in  (fp32 for the scan blend, bf16 for next GEMMs)
    gemm_bf16_wmma<<<ggrid, 256, 0, stream>>>(x_bf, Wi_b, b_in, inp_f, inp_b,
                                              MROWS, DIM, DIM, 0);
    // gate = sigmoid(inp @ W_gate^T + b_gate)
    gemm_bf16_wmma<<<ggrid, 256, 0, stream>>>(inp_b, Wg_b, b_gate, gate_f, nullptr,
                                              MROWS, DIM, DIM, 1);
    // u = inp @ B^T
    gemm_bf16_wmma<<<ggrid, 256, 0, stream>>>(inp_b, Bm_b, nullptr, u_f, nullptr,
                                              MROWS, DIM, DIM, 0);
    // sequential recurrence (persistent, 8 WGPs, A pinned in LDS via TDM)
    scan_wmma<<<8, 256, 0, stream>>>(inp_f, gate_f, u_f, A_b, stateBuf, states_b, bar);
    // out = states @ W_out^T + b_out  (written straight to d_out, fp32)
    gemm_bf16_wmma<<<ggrid, 256, 0, stream>>>(states_b, Wo_b, b_out, out, nullptr,
                                              MROWS, DIM, DIM, 0);
}
